// MLC_22643067584579
// MI455X (gfx1250) — matrix operand order repaired and checked by reading, not compiled, not run on hardware
//
#include <hip/hip_runtime.h>
#include <hip/hip_bf16.h>

#define B_ 4
#define C_ 32
#define H_ 512
#define W_ 512
#define BN_EPS 1e-5f

typedef __attribute__((ext_vector_type(16))) __bf16 v16bf;
typedef __attribute__((ext_vector_type(8)))  __bf16 v8bf;
typedef __attribute__((ext_vector_type(8)))  float  v8f;

__device__ __forceinline__ __bf16 f2bf(float f) {
    unsigned u = __builtin_bit_cast(unsigned, f);
    unsigned r = u + 0x7FFFu + ((u >> 16) & 1u);   // round-to-nearest-even
    unsigned short h = (unsigned short)(r >> 16);
    return __builtin_bit_cast(__bf16, h);
}

// ---------------------------------------------------------------------------
// Kernel 1: conv3x3(C32->C32) + BN + SiLU + conv1x1(C32->1) -> x [B,1,H,W]
// Implicit GEMM via v_wmma_f32_16x16x32_bf16.
// Block = 256 threads (8 waves). Wave w handles 16 consecutive output pixels
// of one row; block covers 128 pixels. M=16 pixels, N=16 couts (x2 acc), K=32
// cins per tap, 9 taps => 18 WMMAs per wave.
// ---------------------------------------------------------------------------
__global__ void MLC_conv_head(const float* __restrict__ cen,
                              const float* __restrict__ w_conv,
                              const float* __restrict__ bn_gamma,
                              const float* __restrict__ bn_beta,
                              const float* __restrict__ bn_mean,
                              const float* __restrict__ bn_var,
                              const float* __restrict__ w_1x1,
                              const float* __restrict__ b_1x1,
                              float* __restrict__ xbuf) {
    // LDS tiles (bf16): activations [3 rows][130 cols][32 cin], weights [9 taps][32 cout][32 cin]
    __shared__ __bf16 sAct[3 * 130 * 32];   // 24960 B
    __shared__ __bf16 sW[9 * 32 * 32];      // 18432 B

    const int tid = threadIdx.x;
    const int b  = blockIdx.z;
    const int h  = blockIdx.y;
    const int w0 = blockIdx.x * 128;

    // --- load weights: sW[tap*1024 + cout*32 + cin] = w_conv[cout][cin][tap]
    for (int i = tid; i < 9 * 32 * 32; i += 256) {
        int tap  = i >> 10;
        int cout = (i >> 5) & 31;
        int cin  = i & 31;
        sW[i] = f2bf(w_conv[(cout * 32 + cin) * 9 + tap]);
    }
    // --- load activation tile with halo, zero-padded (SAME)
    for (int i = tid; i < 3 * 32 * 130; i += 256) {
        int col = i % 130;
        int t   = i / 130;
        int cin = t & 31;
        int row = t >> 5;
        int hh = h + row - 1;
        int ww = w0 + col - 1;
        float v = 0.0f;
        if (hh >= 0 && hh < H_ && ww >= 0 && ww < W_)
            v = cen[((b * C_ + cin) * H_ + hh) * W_ + ww];
        sAct[(row * 130 + col) * 32 + cin] = f2bf(v);
    }
    __syncthreads();

    const int lane  = tid & 31;
    const int wv    = tid >> 5;
    const int hi    = lane >> 4;   // lane half
    const int ln    = lane & 15;
    const int mbase = wv * 16;

    v8f acc0 = {};   // cout 0..15
    v8f acc1 = {};   // cout 16..31

#pragma unroll
    for (int tap = 0; tap < 9; ++tap) {
        const int dy = tap / 3;
        const int dx = tap % 3;

        // A fragment: lane ln holds pixel m=ln; element j -> K = hi*16 + j
        const __bf16* ap = &sAct[((dy * 130) + (mbase + ln + dx)) * 32 + hi * 16];
        v8bf aLo = *(const v8bf*)ap;          // K = hi*16 + 0..7
        v8bf aHi = *(const v8bf*)(ap + 8);    // K = hi*16 + 8..15
        v16bf A;
#pragma unroll
        for (int j = 0; j < 8; ++j) { A[j] = aLo[j]; A[8 + j] = aHi[j]; }

        // B fragments: lane -> N = ln, element j -> K = hi*16 + j (contiguous cin)
        const __bf16* wp0 = &sW[tap * 1024 + ln * 32 + hi * 16];
        const __bf16* wp1 = &sW[tap * 1024 + (16 + ln) * 32 + hi * 16];
        v8bf b0lo = *(const v8bf*)wp0;
        v8bf b0hi = *(const v8bf*)(wp0 + 8);
        v8bf b1lo = *(const v8bf*)wp1;
        v8bf b1hi = *(const v8bf*)(wp1 + 8);
        v16bf B0, B1;
#pragma unroll
        for (int j = 0; j < 8; ++j) {
            B0[j] = b0lo[j]; B0[8 + j] = b0hi[j];
            B1[j] = b1lo[j]; B1[8 + j] = b1hi[j];
        }

        acc0 = __builtin_amdgcn_wmma_f32_16x16x32_bf16(false, A, false, B0,
                                                       (short)0, acc0, false, false);
        acc1 = __builtin_amdgcn_wmma_f32_16x16x32_bf16(false, A, false, B1,
                                                       (short)0, acc1, false, false);
    }

    // Epilogue: BN + SiLU + 1x1 reduction over couts.
    // Lane holds (v=0..7): pixel m = v + 8*hi, cout = ln (acc0) / 16+ln (acc1).
    const float inv0 = bn_gamma[ln]      * rsqrtf(bn_var[ln]      + BN_EPS);
    const float inv1 = bn_gamma[16 + ln] * rsqrtf(bn_var[16 + ln] + BN_EPS);
    const float sh0  = bn_beta[ln]      - bn_mean[ln]      * inv0;
    const float sh1  = bn_beta[16 + ln] - bn_mean[16 + ln] * inv1;
    const float wa0  = w_1x1[ln];
    const float wa1  = w_1x1[16 + ln];
    const float bias = b_1x1[0];

#pragma unroll
    for (int v = 0; v < 8; ++v) {
        float y0 = acc0[v] * inv0 + sh0;
        float y1 = acc1[v] * inv1 + sh1;
        y0 = y0 / (1.0f + __expf(-y0));   // SiLU
        y1 = y1 / (1.0f + __expf(-y1));
        float p = y0 * wa0 + y1 * wa1;
        // reduce across 16-lane half (xor masks stay within the half)
#pragma unroll
        for (int m = 1; m < 16; m <<= 1) p += __shfl_xor(p, m, 32);
        if (ln == 0) {
            int wpix = w0 + mbase + v + 8 * hi;
            xbuf[(b * H_ + h) * W_ + wpix] = p + bias;
        }
    }
}

// ---------------------------------------------------------------------------
// Kernel 2: attention map from x: dilated shifted-diff products, min over 4
// directions, combine over shifts s in {1,3}, sigmoid(relu(.)).
// ---------------------------------------------------------------------------
__global__ void MLC_att(const float* __restrict__ x, float* __restrict__ att) {
    int idx = blockIdx.x * blockDim.x + threadIdx.x;
    if (idx >= B_ * H_ * W_) return;
    int w = idx % W_;
    int h = (idx / W_) % H_;
    int b = idx / (W_ * H_);
    const float* xb = x + b * H_ * W_;
    const float c = xb[h * W_ + w];

    const int dys[4] = {-1, -1, -1, 0};
    const int dxs[4] = {-1,  0,  1, -1};
    float r[2];
#pragma unroll
    for (int si = 0; si < 2; ++si) {
        const int s = si ? 3 : 1;
        float best = 3.402823466e38f;
#pragma unroll
        for (int d = 0; d < 4; ++d) {
            int h1 = h + dys[d] * s, w1 = w + dxs[d] * s;
            int h2 = h - dys[d] * s, w2 = w - dxs[d] * s;
            float xa = (h1 >= 0 && h1 < H_ && w1 >= 0 && w1 < W_) ? xb[h1 * W_ + w1] : 0.0f;
            float xc = (h2 >= 0 && h2 < H_ && w2 >= 0 && w2 < W_) ? xb[h2 * W_ + w2] : 0.0f;
            float o1 = c - xa;   // forward diff conv (k1)
            float o2 = c - xc;   // flipped diff conv (k2)
            best = fminf(best, o1 * o2);
        }
        r[si] = best;
    }
    float a = (fmaxf(r[0], r[1]) + 0.5f * (r[0] + r[1])) * 0.5f;
    a = a > 0.0f ? a : 0.0f;               // relu
    att[idx] = 1.0f / (1.0f + __expf(-a)); // sigmoid
}

// ---------------------------------------------------------------------------
// Kernel 3: out = cen * att (broadcast over channels), float4 streaming.
// ---------------------------------------------------------------------------
__global__ void MLC_mul(const float4* __restrict__ cen4,
                        const float* __restrict__ att,
                        float4* __restrict__ out4) {
    const int total4 = (B_ * C_ * H_ * W_) / 4;
    int idx = blockIdx.x * blockDim.x + threadIdx.x;
    if (idx >= total4) return;
    int e = idx * 4;
    int pix = e % (H_ * W_);
    int b   = e / (C_ * H_ * W_);

    if (idx + 1024 < total4)
        __builtin_prefetch((const float*)&cen4[idx + 1024], 0, 0); // global_prefetch_b8

    float4 cv = cen4[idx];
    float4 av = *(const float4*)(att + b * H_ * W_ + pix);
    float4 ov;
    ov.x = cv.x * av.x;
    ov.y = cv.y * av.y;
    ov.z = cv.z * av.z;
    ov.w = cv.w * av.w;
    out4[idx] = ov;
}

extern "C" void kernel_launch(void* const* d_in, const int* in_sizes, int n_in,
                              void* d_out, int out_size, void* d_ws, size_t ws_size,
                              hipStream_t stream) {
    (void)in_sizes; (void)n_in; (void)out_size; (void)ws_size;
    const float* cen      = (const float*)d_in[0];
    const float* w_conv   = (const float*)d_in[1];
    const float* bn_gamma = (const float*)d_in[2];
    const float* bn_beta  = (const float*)d_in[3];
    const float* bn_mean  = (const float*)d_in[4];
    const float* bn_var   = (const float*)d_in[5];
    const float* w_1x1    = (const float*)d_in[6];
    const float* b_1x1    = (const float*)d_in[7];
    float* out = (float*)d_out;

    float* xbuf   = (float*)d_ws;                 // B*H*W floats = 4 MB
    float* attbuf = xbuf + (size_t)B_ * H_ * W_;  // B*H*W floats = 4 MB

    dim3 g1(W_ / 128, H_, B_);
    MLC_conv_head<<<g1, 256, 0, stream>>>(cen, w_conv, bn_gamma, bn_beta,
                                          bn_mean, bn_var, w_1x1, b_1x1, xbuf);

    int npix = B_ * H_ * W_;
    MLC_att<<<(npix + 255) / 256, 256, 0, stream>>>(xbuf, attbuf);

    int total4 = (B_ * C_ * H_ * W_) / 4;
    MLC_mul<<<(total4 + 255) / 256, 256, 0, stream>>>((const float4*)cen, attbuf,
                                                      (float4*)out);
}
